// MambaLayer_73624329388588
// MI455X (gfx1250) — compile-verified, matrix operand order checked by reading, not run on hardware
//
#include <hip/hip_runtime.h>
#include <math.h>

#define L_TOT   13824
#define NCH_DIM 64          // model channels
#define D_INNER 128
#define D_STATE 16
#define DT_RANK 4
#define NSLICES 24
#define PER     (L_TOT / NSLICES)   // 576
#define MT      (L_TOT / 16)        // 864 M-tiles
#define CHUNK   128
#define NCH     (L_TOT / CHUNK)     // 108 chunks
#define DN      (D_INNER * D_STATE) // 2048 recurrence lanes per branch

typedef __attribute__((ext_vector_type(16))) _Float16 v16h;
typedef __attribute__((ext_vector_type(8)))  float    v8f;
typedef __attribute__((ext_vector_type(2)))  float    v2f;
typedef int v4i_t __attribute__((__vector_size__(16)));   // matches builtin prototype

#if __has_builtin(__builtin_amdgcn_global_load_async_to_lds_b128)
#define HAS_ASYNC_LDS 1
#endif

// Async copy 16B global -> LDS (ASYNCcnt path); sync fallback if builtin absent.
// Prototype (from hipcc diagnostic): (v4i addrspace(1)*, v4i addrspace(3)*, imm, imm)
__device__ __forceinline__ void async_ld_b128(const float* g, float* l) {
#ifdef HAS_ASYNC_LDS
  __builtin_amdgcn_global_load_async_to_lds_b128(
      (__attribute__((address_space(1))) v4i_t*)g,
      (__attribute__((address_space(3))) v4i_t*)l, 0, 0);
#else
  *(float4*)l = *(const float4*)g;
#endif
}
__device__ __forceinline__ void async_wait_all() {
#ifdef HAS_ASYNC_LDS
#if __has_builtin(__builtin_amdgcn_s_wait_asynccnt)
  __builtin_amdgcn_s_wait_asynccnt(0);
#else
  asm volatile("s_wait_asynccnt 0" ::: "memory");
#endif
#endif
}

__device__ __forceinline__ float sigmoidf_fast(float v) {
  return __builtin_amdgcn_rcpf(1.0f + __expf(-v));       // v_rcp_f32, no div chain
}
__device__ __forceinline__ float siluf(float v)     { return v * sigmoidf_fast(v); }
__device__ __forceinline__ float softplusf(float v) {
  return v > 20.0f ? v : __logf(1.0f + __expf(v));       // v_log_f32 + v_exp_f32
}

// Branch sequence mapping: seq position l -> original position (same map for gather & scatter)
__device__ __forceinline__ int map_pos(int b, int l) {
  if (b == 0) return l;
  if (b == 1) return L_TOT - 1 - l;
  return (l % NSLICES) * PER + (l / NSLICES);
}

// Load a 16x32 f16 WMMA fragment (A role: row=m along K; B role: row=n along K of
// row-major W so B[k][n] = W[n][k]). Per ISA 7.12.2: elems 0..7 -> K=k0+8*half+j,
// elems 8..15 -> K=k0+16+8*half+j.
__device__ __forceinline__ v16h load_frag(const _Float16* base, int row, int stride,
                                          int k0, int lane) {
  int half = (lane >> 4) & 1;
  const _Float16* p = base + row * stride + k0 + 8 * half;
  v16h v;
#pragma unroll
  for (int j = 0; j < 8; ++j) v[j] = p[j];
#pragma unroll
  for (int j = 0; j < 8; ++j) v[8 + j] = p[16 + j];
  return v;
}

// ---------------- weight prep: f32 -> f16 row-major (x_proj padded 36->48 rows) ---------
__global__ void k_prep_weights(const float* __restrict__ Win, const float* __restrict__ Wx,
                               const float* __restrict__ Wout, _Float16* __restrict__ fWin,
                               _Float16* __restrict__ fWx, _Float16* __restrict__ fWout) {
  int stride = gridDim.x * blockDim.x;
  int idx = blockIdx.x * blockDim.x + threadIdx.x;
  for (int i = idx; i < 256 * 64; i += stride) fWin[i] = (_Float16)Win[i];
  for (int i = idx; i < 3 * 48 * 128; i += stride) {
    int b = i / (48 * 128), r = (i / 128) % 48, k = i & 127;
    fWx[i] = (_Float16)((r < 36) ? Wx[(b * 36 + r) * 128 + k] : 0.0f);
  }
  for (int i = idx; i < 64 * 128; i += stride) fWout[i] = (_Float16)Wout[i];
}

// ---------------- LayerNorm + in_proj (WMMA), writes xs and silu(z) --------------------
__global__ void k_ln_inproj(const float* __restrict__ x, const float* __restrict__ lnw,
                            const float* __restrict__ lnb, const _Float16* __restrict__ fWin,
                            float* __restrict__ xs, float* __restrict__ zs) {
  __shared__ float    xt[NCH_DIM][16];   // [c][r] so each 16B global chunk lands contiguous
  __shared__ _Float16 xh[16][NCH_DIM];   // [r][c] row-major for A fragments
  __shared__ float mu_s[16], rs_s[16];
  int t = threadIdx.x;
  int l0 = blockIdx.x * 16;
  {
    int c = t >> 2, q = t & 3;           // 256 threads x 16B = whole 16x64 tile, one op each
    async_ld_b128(x + (size_t)c * L_TOT + l0 + q * 4, &xt[c][q * 4]);
    async_wait_all();
  }
  __syncthreads();
  if (t < 16) {
    float s = 0.f, s2 = 0.f;
    for (int c = 0; c < NCH_DIM; ++c) { float v = xt[c][t]; s += v; s2 += v * v; }
    float mu = s / NCH_DIM;
    mu_s[t] = mu;
    rs_s[t] = rsqrtf(s2 / NCH_DIM - mu * mu + 1e-5f);
  }
  __syncthreads();
  int r = t & 15, cb = t >> 4;
#pragma unroll
  for (int j = 0; j < 4; ++j) {
    int c = cb + 16 * j;
    xh[r][c] = (_Float16)((xt[c][r] - mu_s[r]) * rs_s[r] * lnw[c] + lnb[c]);
  }
  __syncthreads();
  int wave = t >> 5, lane = t & 31, m = lane & 15, half = lane >> 4;
  for (int nt = wave * 2; nt < wave * 2 + 2; ++nt) {      // 8 waves x 2 = 16 N-tiles (256 cols)
    v8f acc = {};
#pragma unroll
    for (int kk = 0; kk < NCH_DIM; kk += 32) {
      v16h a  = load_frag(&xh[0][0], m, NCH_DIM, kk, lane);
      v16h bf = load_frag(fWin, nt * 16 + m, NCH_DIM, kk, lane);
      acc = __builtin_amdgcn_wmma_f32_16x16x32_f16(false, a, false, bf, (short)0, acc, false, false);
    }
    // Branchless: cndmask-select destination + silu for z columns (no exec divergence)
    bool isz = nt >= 8;
    float* dst = isz ? zs : xs;
    int col = (nt * 16 + m) & (D_INNER - 1);
#pragma unroll
    for (int rr = 0; rr < 8; ++rr) {
      float v = acc[rr];
      float sv = siluf(v);
      dst[(size_t)(l0 + rr + 8 * half) * D_INNER + col] = isz ? sv : v;
    }
  }
}

// ---------------- per-branch: causal conv4+SiLU, x_proj (WMMA f16), dt (WMMA f32 K=4) ---
__global__ void k_branch_prep(const float* __restrict__ xs, const float* __restrict__ conv_w,
                              const float* __restrict__ conv_b, const _Float16* __restrict__ fWx,
                              const float* __restrict__ dtw, const float* __restrict__ dtb,
                              float* __restrict__ xc_ws, float* __restrict__ dt_ws,
                              float* __restrict__ bc_ws, float* __restrict__ cc_ws) {
  __shared__ float    xcf[16][D_INNER];
  __shared__ _Float16 xch[16][D_INNER];
  __shared__ float    proj[16][48];
  int t = threadIdx.x;
  int l0 = blockIdx.x * 16;
  int br = blockIdx.y;
  for (int i = t; i < 16 * D_INNER; i += 256) {
    int r = i >> 7, d = i & 127, l = l0 + r;
    float acc = conv_b[br * D_INNER + d];
    const float* wd = conv_w + (size_t)(br * D_INNER + d) * 4;
#pragma unroll
    for (int j = 0; j < 4; ++j) {
      int sp = l - 3 + j;
      if (sp >= 0) acc += wd[j] * xs[(size_t)map_pos(br, sp) * D_INNER + d];
    }
    float v = siluf(acc);
    xcf[r][d] = v;
    xch[r][d] = (_Float16)v;
  }
  __syncthreads();
  int wave = t >> 5, lane = t & 31, m = lane & 15, half = lane >> 4;
  if (wave < 3) {                                   // 3 N-tiles cover 48 (36 used) cols
    v8f acc = {};
    const _Float16* Wb = fWx + (size_t)br * 48 * D_INNER;
#pragma unroll
    for (int kk = 0; kk < D_INNER; kk += 32) {
      v16h a  = load_frag(&xch[0][0], m, D_INNER, kk, lane);
      v16h bf = load_frag(Wb, wave * 16 + m, D_INNER, kk, lane);
      acc = __builtin_amdgcn_wmma_f32_16x16x32_f16(false, a, false, bf, (short)0, acc, false, false);
    }
#pragma unroll
    for (int rr = 0; rr < 8; ++rr) proj[rr + 8 * half][wave * 16 + m] = acc[rr];
  }
  __syncthreads();
  const float* dwb = dtw + (size_t)br * D_INNER * DT_RANK;
  const float* dbb = dtb + (size_t)br * D_INNER;
  size_t boff = (size_t)br * L_TOT * D_INNER;
  {
    // dt_pre(16x128) = proj[:, :4] (16x4) @ dw^T (4x128), exact f32 WMMA, one N-tile/wave.
    // 32-bit A 16x4 layout: lane m, VGPR j holds K = 2*half + j. B mirrored from dw rows.
    v2f a, b;
    a[0] = proj[m][2 * half + 0];
    a[1] = proj[m][2 * half + 1];
    int d = wave * 16 + m;                         // N role -> inner channel
    b[0] = dwb[d * DT_RANK + 2 * half + 0];
    b[1] = dwb[d * DT_RANK + 2 * half + 1];
    v8f acc = {};
    acc = __builtin_amdgcn_wmma_f32_16x16x4_f32(false, a, false, b, (short)0, acc, false, false);
    float bias = dbb[d];
#pragma unroll
    for (int rr = 0; rr < 8; ++rr) {
      int l = l0 + rr + 8 * half;
      dt_ws[boff + (size_t)l * D_INNER + d] = softplusf(acc[rr] + bias);
    }
  }
  for (int i = t; i < 16 * D_INNER; i += 256) {
    int r = i >> 7, d = i & 127, l = l0 + r;
    xc_ws[boff + (size_t)l * D_INNER + d] = xcf[r][d];
  }
  size_t soff = (size_t)br * L_TOT * D_STATE;
  if (t < 16 * D_STATE) {
    int r = t >> 4, n = t & 15, l = l0 + r;
    bc_ws[soff + (size_t)l * D_STATE + n] = proj[r][DT_RANK + n];
    cc_ws[soff + (size_t)l * D_STATE + n] = proj[r][DT_RANK + D_STATE + n];
  }
}

// ---------------- chunked scan pass 1: per-chunk (prod a, local end state) --------------
__global__ void k_chunk_scan(const float* __restrict__ dt_ws, const float* __restrict__ xc_ws,
                             const float* __restrict__ bc_ws, const float* __restrict__ a_log,
                             float* __restrict__ Pb, float* __restrict__ Sb) {
  int dn = blockIdx.x * 256 + threadIdx.x;
  int c = blockIdx.y, br = blockIdx.z;
  int d = dn >> 4, n = dn & 15;
  float A = -__expf(a_log[br * DN + dn]);
  const float* dtb = dt_ws + (size_t)br * L_TOT * D_INNER;
  const float* xcb = xc_ws + (size_t)br * L_TOT * D_INNER;
  const float* bcb = bc_ws + (size_t)br * L_TOT * D_STATE;
  float P = 1.0f, S = 0.0f;
  int l = c * CHUNK;
  for (int i = 0; i < CHUNK; ++i, ++l) {
    int lp = l + 8; lp = lp < L_TOT ? lp : L_TOT - 1;     // branchless clamped prefetch
    __builtin_prefetch(&dtb[(size_t)lp * D_INNER + d], 0, 1);
    __builtin_prefetch(&xcb[(size_t)lp * D_INNER + d], 0, 1);
    float dtv = dtb[(size_t)l * D_INNER + d];
    float a = __expf(dtv * A);
    float bv = dtv * xcb[(size_t)l * D_INNER + d] * bcb[(size_t)l * D_STATE + n];
    P *= a;
    S = a * S + bv;
  }
  size_t o = (size_t)(br * NCH + c) * DN + dn;
  Pb[o] = P;
  Sb[o] = S;
}

// ---------------- chunked scan pass 2: prefix over chunks ------------------------------
__global__ void k_chunk_prefix(const float* __restrict__ Pb, const float* __restrict__ Sb,
                               float* __restrict__ Hin) {
  int id = blockIdx.x * 256 + threadIdx.x;
  if (id >= 3 * DN) return;
  int br = id / DN, dn = id % DN;
  float h = 0.0f;
  for (int c = 0; c < NCH; ++c) {
    size_t o = (size_t)(br * NCH + c) * DN + dn;
    Hin[o] = h;
    h = Pb[o] * h + Sb[o];
  }
}

// ---------------- finalize: recompute within chunk, y = <h,C> + xc*D, scatter-add -------
__global__ void k_finalize(const float* __restrict__ dt_ws, const float* __restrict__ xc_ws,
                           const float* __restrict__ bc_ws, const float* __restrict__ cc_ws,
                           const float* __restrict__ a_log, const float* __restrict__ d_ssm,
                           const float* __restrict__ Hin, float* __restrict__ ysum) {
  int t = threadIdx.x;
  int wave = t >> 5, lane = t & 31, half = lane >> 4, n = lane & 15;
  int c = blockIdx.y, br = blockIdx.z;
  int d = blockIdx.x * 16 + wave * 2 + half;          // lanes 0-15: d0, lanes 16-31: d0+1
  float A  = -__expf(a_log[br * DN + d * D_STATE + n]);
  float dp = d_ssm[br * D_INNER + d];
  const float* dtb = dt_ws + (size_t)br * L_TOT * D_INNER;
  const float* xcb = xc_ws + (size_t)br * L_TOT * D_INNER;
  const float* bcb = bc_ws + (size_t)br * L_TOT * D_STATE;
  const float* ccb = cc_ws + (size_t)br * L_TOT * D_STATE;
  float h = Hin[(size_t)(br * NCH + c) * DN + d * D_STATE + n];
  int l = c * CHUNK;
  for (int i = 0; i < CHUNK; ++i, ++l) {
    int lp = l + 8; lp = lp < L_TOT ? lp : L_TOT - 1;
    __builtin_prefetch(&dtb[(size_t)lp * D_INNER + d], 0, 1);
    __builtin_prefetch(&ccb[(size_t)lp * D_STATE + n], 0, 1);
    float dtv = dtb[(size_t)l * D_INNER + d];
    float xcv = xcb[(size_t)l * D_INNER + d];
    float a = __expf(dtv * A);
    h = a * h + dtv * xcv * bcb[(size_t)l * D_STATE + n];
    float val = h * ccb[(size_t)l * D_STATE + n];
#pragma unroll
    for (int m = 8; m >= 1; m >>= 1) val += __shfl_xor(val, m);   // reduce 16 states
    if (n == 0) atomicAdd(&ysum[(size_t)map_pos(br, l) * D_INNER + d], val + xcv * dp);
  }
}

// ---------------- out_proj (WMMA) + residual + channel sums for ECA --------------------
__global__ void k_outproj(const float* __restrict__ ysum, const float* __restrict__ zs,
                          const _Float16* __restrict__ fWout, const float* __restrict__ x,
                          float* __restrict__ out, float* __restrict__ gsum) {
  __shared__ float    yf[16 * D_INNER];
  __shared__ float    zf[16 * D_INNER];
  __shared__ _Float16 yh[16][D_INNER];
  int t = threadIdx.x;                                  // 128 threads, 4 waves
  int l0 = blockIdx.x * 16;
  const float* ysrc = ysum + (size_t)l0 * D_INNER;      // fully contiguous 8KB tiles
  const float* zsrc = zs + (size_t)l0 * D_INNER;
#pragma unroll
  for (int i = 0; i < 4; ++i) {
    int fi = (t + i * 128) * 4;
    async_ld_b128(ysrc + fi, yf + fi);
    async_ld_b128(zsrc + fi, zf + fi);
  }
  async_wait_all();
  __syncthreads();
  _Float16* yhf = &yh[0][0];
  for (int i = t; i < 16 * D_INNER; i += 128) yhf[i] = (_Float16)(yf[i] * zf[i]);
  __syncthreads();
  int wave = t >> 5, lane = t & 31, m = lane & 15, half = lane >> 4;
  v8f acc = {};
#pragma unroll
  for (int kk = 0; kk < D_INNER; kk += 32) {
    v16h a  = load_frag(yhf, m, D_INNER, kk, lane);
    v16h bf = load_frag(fWout, wave * 16 + m, D_INNER, kk, lane);
    acc = __builtin_amdgcn_wmma_f32_16x16x32_f16(false, a, false, bf, (short)0, acc, false, false);
  }
  // Lane owns channel cch; its 8 outputs are contiguous along l -> two b128 stores.
  int cch = wave * 16 + m;
  size_t base = (size_t)cch * L_TOT + l0 + 8 * half;
  float4 x0 = *(const float4*)(x + base);
  float4 x1 = *(const float4*)(x + base + 4);
  float4 o0, o1;
  o0.x = acc[0] + x0.x; o0.y = acc[1] + x0.y; o0.z = acc[2] + x0.z; o0.w = acc[3] + x0.w;
  o1.x = acc[4] + x1.x; o1.y = acc[5] + x1.y; o1.z = acc[6] + x1.z; o1.w = acc[7] + x1.w;
  *(float4*)(out + base)     = o0;
  *(float4*)(out + base + 4) = o1;
  float psum = o0.x + o0.y + o0.z + o0.w + o1.x + o1.y + o1.z + o1.w;
  psum += __shfl_xor(psum, 16);                        // fold half=1 into half=0
  if (half == 0) atomicAdd(&gsum[cch], psum);
}

// ---------------- ECA attention + in-place scale (float4) ------------------------------
__global__ void k_eca_scale(const float* __restrict__ gsum, const float* __restrict__ eca,
                            float* __restrict__ out) {
  int cch = blockIdx.y;
  int i = blockIdx.x * 128 + threadIdx.x;               // float4 index, L/4 = 3456 per chan
  float inv = 1.0f / (float)L_TOT;
  float gm = (cch > 0)           ? gsum[cch - 1] * inv : 0.0f;
  float g0 = gsum[cch] * inv;
  float gp = (cch < NCH_DIM - 1) ? gsum[cch + 1] * inv : 0.0f;
  float att = sigmoidf_fast(gm * eca[0] + g0 * eca[1] + gp * eca[2]);
  float4* o = (float4*)(out + (size_t)cch * L_TOT) + i;
  float4 v = *o;
  v.x *= att; v.y *= att; v.z *= att; v.w *= att;
  *o = v;
}

// ---------------- workspace layout -----------------------------------------------------
static constexpr size_t AL(size_t v) { return (v + 255) & ~(size_t)255; }
static constexpr size_t OFF_WIN  = 0;
static constexpr size_t OFF_WX   = AL(OFF_WIN  + (size_t)256 * 64 * 2);
static constexpr size_t OFF_WOUT = AL(OFF_WX   + (size_t)3 * 48 * 128 * 2);
static constexpr size_t OFF_XS   = AL(OFF_WOUT + (size_t)64 * 128 * 2);
static constexpr size_t OFF_ZS   = AL(OFF_XS   + (size_t)L_TOT * 128 * 4);
static constexpr size_t OFF_XC   = AL(OFF_ZS   + (size_t)L_TOT * 128 * 4);
static constexpr size_t OFF_DT   = AL(OFF_XC   + (size_t)3 * L_TOT * 128 * 4);
static constexpr size_t OFF_BC   = AL(OFF_DT   + (size_t)3 * L_TOT * 128 * 4);
static constexpr size_t OFF_CC   = AL(OFF_BC   + (size_t)3 * L_TOT * 16 * 4);
static constexpr size_t OFF_P    = AL(OFF_CC   + (size_t)3 * L_TOT * 16 * 4);
static constexpr size_t OFF_S    = AL(OFF_P    + (size_t)3 * NCH * DN * 4);
static constexpr size_t OFF_HIN  = AL(OFF_S    + (size_t)3 * NCH * DN * 4);
static constexpr size_t OFF_YSUM = AL(OFF_HIN  + (size_t)3 * NCH * DN * 4);
static constexpr size_t OFF_GSUM = OFF_YSUM + (size_t)L_TOT * 128 * 4;  // adjacent: one memset

extern "C" void kernel_launch(void* const* d_in, const int* in_sizes, int n_in,
                              void* d_out, int out_size, void* d_ws, size_t ws_size,
                              hipStream_t stream) {
  const float* x     = (const float*)d_in[0];
  const float* lnw   = (const float*)d_in[1];
  const float* lnb   = (const float*)d_in[2];
  const float* Win   = (const float*)d_in[3];
  const float* convw = (const float*)d_in[4];
  const float* convb = (const float*)d_in[5];
  const float* Wx    = (const float*)d_in[6];
  const float* dtw   = (const float*)d_in[7];
  const float* dtb   = (const float*)d_in[8];
  const float* alog  = (const float*)d_in[9];
  const float* dssm  = (const float*)d_in[10];
  const float* Wout  = (const float*)d_in[11];
  const float* eca   = (const float*)d_in[12];
  float* out = (float*)d_out;

  char* ws = (char*)d_ws;
  _Float16* fWin  = (_Float16*)(ws + OFF_WIN);
  _Float16* fWx   = (_Float16*)(ws + OFF_WX);
  _Float16* fWout = (_Float16*)(ws + OFF_WOUT);
  float* xs    = (float*)(ws + OFF_XS);
  float* zs    = (float*)(ws + OFF_ZS);
  float* xc_ws = (float*)(ws + OFF_XC);
  float* dt_ws = (float*)(ws + OFF_DT);
  float* bc_ws = (float*)(ws + OFF_BC);
  float* cc_ws = (float*)(ws + OFF_CC);
  float* Pb    = (float*)(ws + OFF_P);
  float* Sb    = (float*)(ws + OFF_S);
  float* Hin   = (float*)(ws + OFF_HIN);
  float* ysum  = (float*)(ws + OFF_YSUM);
  float* gsum  = (float*)(ws + OFF_GSUM);

  (void)hipMemsetAsync(ysum, 0, (size_t)L_TOT * 128 * 4 + 256, stream);  // ysum + gsum

  k_prep_weights<<<64, 256, 0, stream>>>(Win, Wx, Wout, fWin, fWx, fWout);
  k_ln_inproj<<<MT, 256, 0, stream>>>(x, lnw, lnb, fWin, xs, zs);
  k_branch_prep<<<dim3(MT, 3), 256, 0, stream>>>(xs, convw, convb, fWx, dtw, dtb,
                                                 xc_ws, dt_ws, bc_ws, cc_ws);
  k_chunk_scan<<<dim3(DN / 256, NCH, 3), 256, 0, stream>>>(dt_ws, xc_ws, bc_ws, alog, Pb, Sb);
  k_chunk_prefix<<<24, 256, 0, stream>>>(Pb, Sb, Hin);
  k_finalize<<<dim3(D_INNER / 16, NCH, 3), 256, 0, stream>>>(dt_ws, xc_ws, bc_ws, cc_ws,
                                                             alog, dssm, Hin, ysum);
  k_outproj<<<MT, 128, 0, stream>>>(ysum, zs, fWout, x, out, gsum);
  k_eca_scale<<<dim3(L_TOT / 4 / 128, NCH_DIM), 128, 0, stream>>>(gsum, eca, out);
}
